// Model_17085379903564
// MI455X (gfx1250) — compile-verified, matrix operand order checked by reading, not run on hardware
//
#include <hip/hip_runtime.h>

typedef unsigned short u16;
typedef __attribute__((ext_vector_type(16))) __bf16 v16bf;
typedef __attribute__((ext_vector_type(8)))  __bf16 v8bf;
typedef __attribute__((ext_vector_type(8)))  float  v8f;
typedef __attribute__((ext_vector_type(2)))  float  v2f;

#define NORD   256
#define MODES  32
#define BE     128          // B_SZ(4) * E_IN(32)
#define KCOLS  16384        // NORD * MODES * 2 (re/im interleaved)
#define TSTAR  511          // PRED_LEN - 1
#define MT     4            // M-tiles per wave in the bf16 WMMA GEMM

__device__ __forceinline__ u16 f2bf(float f) {           // round-to-nearest-even fp32->bf16
  unsigned u = __float_as_uint(f);
  u += 0x7FFFu + ((u >> 16) & 1u);
  return (u16)(u >> 16);
}

// ---------------- RevIN stats + normalized, sliced input rows (bf16) ----------------
__global__ void stats_f_kernel(const float* __restrict__ x,
                               float* __restrict__ meanv, float* __restrict__ stdv,
                               u16* __restrict__ f0, u16* __restrict__ f1) {
  int be = blockIdx.x;                 // 0..127
  int b = be >> 5, e = be & 31;
  int tid = threadIdx.x;               // 256 threads
  float s = 0.f, s2 = 0.f;
  for (int t = tid; t < 1024; t += 256) {
    float v = x[(b * 1024 + t) * 32 + e];
    s += v; s2 += v * v;
  }
  __shared__ float sh[256], sh2[256];
  __shared__ float smean, sinv;
  sh[tid] = s; sh2[tid] = s2;
  __syncthreads();
  for (int o = 128; o > 0; o >>= 1) {
    if (tid < o) { sh[tid] += sh[tid + o]; sh2[tid] += sh2[tid + o]; }
    __syncthreads();
  }
  if (tid == 0) {
    float m = sh[0] * (1.f / 1024.f);
    float var = sh2[0] * (1.f / 1024.f) - m * m;
    float sd = sqrtf(var + 1e-5f);
    smean = m; sinv = 1.f / sd;
    meanv[be] = m; stdv[be] = sd;
  }
  __syncthreads();
  float m = smean, inv = sinv;
  for (int k = tid; k < 512; k += 256) {          // scale 0: last 512 steps
    float v = x[(b * 1024 + (512 + k)) * 32 + e];
    f0[be * 512 + k] = f2bf((v - m) * inv);
  }
  for (int k = tid; k < 1024; k += 256) {         // scale 1: all 1024 steps
    float v = x[(b * 1024 + k) * 32 + e];
    f1[be * 1024 + k] = f2bf((v - m) * inv);
  }
}

// ---------------- LMU transition + bilinear discretization ----------------
__device__ __forceinline__ float lmu_a(int r, int c) {   // A[r][c]
  float m = 2.f * r + 1.f;
  if (r < c) return -m;
  return ((r - c) & 1) ? m : -m;                          // (-1)^(r-c+1) * (2r+1)
}

// Augmented [ M | I + dt/2 A | dt*Bv ], M = I - dt/2 A   (256 x 513)
__global__ void build_aug_kernel(float* __restrict__ aug, float dt) {
  int idx = blockIdx.x * blockDim.x + threadIdx.x;
  if (idx >= 256 * 513) return;
  int r = idx / 513, c = idx % 513;
  float v;
  if (c < 256) {
    v = (r == c ? 1.f : 0.f) - 0.5f * dt * lmu_a(r, c);
  } else if (c < 512) {
    int ca = c - 256;
    v = (r == ca ? 1.f : 0.f) + 0.5f * dt * lmu_a(r, ca);
  } else {
    float m = 2.f * r + 1.f;
    v = dt * ((r & 1) ? -m : m);                          // dt * (-1)^r (2r+1)
  }
  aug[idx] = v;
}

// Gauss-Jordan: left block -> I, then cols[256:512]=Ad, col 512=Bd.
// (M is well-conditioned for these dt; no pivoting needed.)  1 block, 1024 thr.
__global__ void gauss_jordan_kernel(float* __restrict__ aug) {
  int tid = threadIdx.x;
  __shared__ float piv[513];
  __shared__ float inv;
  for (int p = 0; p < 256; ++p) {
    if (tid == 0) inv = 1.f / aug[p * 513 + p];
    __syncthreads();
    for (int c = tid; c < 513; c += 1024) {
      float v = aug[p * 513 + c] * inv;
      piv[c] = v;
      aug[p * 513 + c] = v;
    }
    __syncthreads();
    int r = tid >> 2, sub = tid & 3;                      // 4 threads per row
    float f = (r != p) ? aug[r * 513 + p] : 0.f;
    __syncthreads();
    if (r != p) {
      for (int c = p + sub; c < 513; c += 4)
        aug[r * 513 + c] -= f * piv[c];
    }
    __syncthreads();
  }
}

// Ad (row major), Q1 = Ad^T, K[0] = Bd
__global__ void extract_kernel(const float* __restrict__ aug, float* __restrict__ adm,
                               float* __restrict__ q1, float* __restrict__ kseq) {
  int r = blockIdx.x, c = threadIdx.x;
  float v = aug[r * 513 + 256 + c];
  adm[r * 256 + c] = v;
  q1[c * 256 + r] = v;
  if (c == 0) kseq[r] = aug[r * 513 + 512];
}

// K[j] = Ad * K[j-1], j = 1..31 (serial seed; rest done by doubling GEMMs)
__global__ void seed_kernel(const float* __restrict__ adm, float* __restrict__ kseq) {
  int t = threadIdx.x;                                    // 256
  __shared__ float kv[256];
  kv[t] = kseq[t];
  __syncthreads();
  for (int j = 1; j < 32; ++j) {
    float s = 0.f;
    const float* row = adm + t * 256;
    for (int c = 0; c < 256; ++c) s = fmaf(row[c], kv[c], s);
    __syncthreads();
    kv[t] = s;
    kseq[j * 256 + t] = s;
    __syncthreads();
  }
}

// fp32 NN GEMM on the matrix pipe: V_WMMA_F32_16X16X4_F32 (exact f32 accumulate).
// One wave per block; dims multiples of 16.  Used only for the Ad-power chain.
__global__ void gemm_nn_f32_wmma(const float* __restrict__ A, const float* __restrict__ B,
                                 float* __restrict__ C, int M, int N, int K) {
  int lane = threadIdx.x;                                 // 32
  int row = blockIdx.y * 16 + (lane & 15);
  int col = blockIdx.x * 16 + (lane & 15);
  int kh = (lane < 16) ? 0 : 2;                           // f32 A/B lane k-sets: {0,1}/{2,3}
  const float* arow = A + (size_t)row * K + kh;
  const float* bcol = B + col;
  v8f acc = {};
  for (int kb = 0; kb < K; kb += 4) {
    v2f a = *(const v2f*)(arow + kb);                     // A[row][kb+kh .. +1]
    v2f b;
    b.x = bcol[(size_t)(kb + kh) * N];                    // B[kb+kh][col]
    b.y = bcol[(size_t)(kb + kh + 1) * N];
    acc = __builtin_amdgcn_wmma_f32_16x16x4_f32(false, a, false, b,
                                                (short)0, acc, false, false);
  }
  int rbase = blockIdx.y * 16 + ((lane >> 4) & 1) * 8;
  int cn = blockIdx.x * 16 + (lane & 15);
#pragma unroll
  for (int v = 0; v < 8; ++v)
    C[(size_t)(rbase + v) * N + cn] = acc[v];
}

// Em rows L-512..L-1 (Legendre three-term recurrence), bf16, layout [l][n]
__global__ void em_kernel(u16* __restrict__ em, int L) {
  int l = blockIdx.x * blockDim.x + threadIdx.x;
  if (l >= 512) return;
  int tg = L - 512 + l;
  float x = 1.f - 2.f * (float)tg / (float)L;
  float p0 = 1.f, p1 = x;
  u16* row = em + (size_t)l * 256;
  row[0] = f2bf(p0);
  row[1] = f2bf(p1);
  for (int n = 1; n < 255; ++n) {
    float p2 = ((2.f * n + 1.f) * x * p1 - (float)n * p0) / (float)(n + 1);
    row[n + 1] = f2bf(p2);
    p0 = p1; p1 = p2;
  }
}

// G^T: row (i*32+x)*2+comp, length L.  G[k,i,x] = e^{-iwk} * P[L-1-k,i,x]
__global__ void gscan_kernel(const float* __restrict__ kseq, u16* __restrict__ gt, int L) {
  int x = blockIdx.x;                                     // 0..31 (one mode per block)
  int i = threadIdx.x;                                    // 0..255 (coalesced K reads)
  float omega = 6.2831853071795864f * (float)x / (float)L;
  u16* gre = gt + ((size_t)(i * 32 + x) * 2) * L;
  u16* gim = gre + L;
  float ar = 0.f, ai = 0.f;
  for (int j = 0; j < L; ++j) {
    float kv = kseq[j * 256 + i];
    float sj, cj; __sincosf(omega * (float)j, &sj, &cj);  // e^{-iwj} = c - i s
    ar = fmaf(kv, cj, ar);
    ai = fmaf(-kv, sj, ai);
    int k = L - 1 - j;
    float sk, ck; __sincosf(omega * (float)k, &sk, &ck);
    gre[k] = f2bf(ar * ck + ai * sk);
    gim[k] = f2bf(ai * ck - ar * sk);
  }
}

// W-hat^T: [o][(i*32+x)*2+comp], folds irfft-at-t* twiddles + 1/L + mode weights into w
__global__ void wt_kernel(const float* __restrict__ wre, const float* __restrict__ wim,
                          u16* __restrict__ wt, int s, int L) {
  int idx = blockIdx.x * blockDim.x + threadIdx.x;        // 65536
  int o = idx & 255, i = idx >> 8;
  const float* pr = wre + (((size_t)s * 256 + i) * 256 + o) * 32;
  const float* pi = wim + (((size_t)s * 256 + i) * 256 + o) * 32;
  u16* wrow = wt + (size_t)o * KCOLS + (size_t)i * 64;
  float invL = 1.f / (float)L;
  for (int xm = 0; xm < 32; ++xm) {
    float th = 6.2831853071795864f * (float)xm * (float)TSTAR / (float)L;
    float sx, cx; __sincosf(th, &sx, &cx);
    float g = (xm == 0) ? 1.f : 2.f;
    float cc = g * cx * invL, ss = g * sx * invL;
    float wr = pr[xm], wi = pi[xm];
    wrow[xm * 2 + 0] = f2bf(cc * wr - ss * wi);           // multiplies a_re
    wrow[xm * 2 + 1] = f2bf(-(cc * wi + ss * wr));        // multiplies a_im
  }
}

// ---------------- bf16 WMMA GEMM, C = A(MxK) * Bt(NxK)^T, f32 accumulate ----------------
// B fragment is staged LDS-side with double-buffered global_load_async_to_lds_b128,
// gated by s_wait_asynccnt; one B fragment feeds MT=4 WMMAs (4 M-tiles per wave).
__device__ __forceinline__ v16bf load_frag16(const u16* p) {
  v8bf lo = *(const v8bf*)(p);                            // k: +0..7   (rel. to lane half)
  v8bf hi = *(const v8bf*)(p + 16);                       // k: +16..23
  return __builtin_shufflevector(lo, hi, 0,1,2,3,4,5,6,7,8,9,10,11,12,13,14,15);
}

__device__ __forceinline__ void async_b128(unsigned ldsoff, const u16* g) {
  asm volatile("global_load_async_to_lds_b128 %0, %1, off"
               :: "v"(ldsoff), "v"(g) : "memory");
}

__global__ void gemm_bf16_nt(const u16* __restrict__ A, const u16* __restrict__ Bt,
                             void* __restrict__ C, int M, int N, int K, int out_bf16) {
  __shared__ __align__(16) u16 bbuf[2][32 * 16];          // 2 x 1KB: lane slot = 32B
  int lane = threadIdx.x;                                 // one wave32 per block
  int n0 = blockIdx.x * 16;
  int m0 = blockIdx.y * (16 * MT);
  int khalf = (lane < 16) ? 0 : 8;                        // ISA 16-bit A/B lane k-sets
  const u16* brow = Bt + (size_t)(n0 + (lane & 15)) * K + khalf;
  const u16* arow = A + (size_t)(m0 + (lane & 15)) * K + khalf;
  unsigned lds0 = (unsigned)(size_t)&bbuf[0][lane * 16];  // low 32 bits = LDS offset
  unsigned lds1 = (unsigned)(size_t)&bbuf[1][lane * 16];

  // stage kb=0 B chunks (elements +0..7 and +16..23 of this lane's k-window)
  async_b128(lds0, brow);
  async_b128(lds0 + 16, brow + 16);

  v8f acc[MT] = {};
  int buf = 0;
  for (int kb = 0; kb < K; kb += 32) {
    if (kb + 32 < K) {                                    // uniform branch: EXEC stays full
      unsigned nlds = (buf ? lds0 : lds1);
      const u16* nx = brow + kb + 32;
      async_b128(nlds, nx);
      async_b128(nlds + 16, nx + 16);
      asm volatile("s_wait_asynccnt 0x2" ::: "memory");   // current buffer ready
    } else {
      asm volatile("s_wait_asynccnt 0x0" ::: "memory");
    }
    v8bf blo = *(const v8bf*)&bbuf[buf][lane * 16];       // ds_load_b128
    v8bf bhi = *(const v8bf*)&bbuf[buf][lane * 16 + 8];
    v16bf bf = __builtin_shufflevector(blo, bhi,
                                       0,1,2,3,4,5,6,7,8,9,10,11,12,13,14,15);
#pragma unroll
    for (int t = 0; t < MT; ++t) {
      const u16* ap = arow + (size_t)t * 16 * K + kb;
      if (kb + 128 < K) __builtin_prefetch(ap + 128, 0, 1);
      v16bf af = load_frag16(ap);
      acc[t] = __builtin_amdgcn_wmma_f32_16x16x32_bf16(false, af, false, bf,
                                                       (short)0, acc[t], false, false);
    }
    buf ^= 1;
  }

  int cn = n0 + (lane & 15);
  int rsub = ((lane >> 4) & 1) * 8;                       // D: VGPR v -> M = v (+8 hi lanes)
#pragma unroll
  for (int t = 0; t < MT; ++t) {
    int rbase = m0 + t * 16 + rsub;
    if (out_bf16) {
      u16* Cb = (u16*)C;
#pragma unroll
      for (int v = 0; v < 8; ++v) Cb[(size_t)(rbase + v) * N + cn] = f2bf(acc[t][v]);
    } else {
      float* Cf = (float*)C;
#pragma unroll
      for (int v = 0; v < 8; ++v) Cf[(size_t)(rbase + v) * N + cn] = acc[t][v];
    }
  }
}

// ---------------- scale combine + de-normalize ----------------
__global__ void final_kernel(const float* __restrict__ dec0, const float* __restrict__ dec1,
                             const float* __restrict__ mlpw, const float* __restrict__ mlpb,
                             const float* __restrict__ meanv, const float* __restrict__ stdv,
                             float* __restrict__ out) {
  int idx = blockIdx.x * blockDim.x + threadIdx.x;        // 4*512*32 = 65536
  int e = idx & 31;
  int l = (idx >> 5) & 511;
  int b = idx >> 14;
  int be = b * 32 + e;
  float v = dec0[(size_t)be * 512 + l] * mlpw[0]
          + dec1[(size_t)be * 512 + l] * mlpw[1] + mlpb[0];
  out[idx] = v * stdv[be] + meanv[be];
}

extern "C" void kernel_launch(void* const* d_in, const int* in_sizes, int n_in,
                              void* d_out, int out_size, void* d_ws, size_t ws_size,
                              hipStream_t stream) {
  (void)in_sizes; (void)n_in; (void)out_size; (void)ws_size;
  const float* x_enc = (const float*)d_in[0];
  const float* wre   = (const float*)d_in[1];
  const float* wim   = (const float*)d_in[2];
  const float* mlpw  = (const float*)d_in[3];
  const float* mlpb  = (const float*)d_in[4];
  float* out = (float*)d_out;

  char* p = (char*)d_ws;
  auto carve = [&](size_t bytes) -> void* {
    void* r = (void*)p; p += (bytes + 255) & ~(size_t)255; return r;
  };
  float* aug   = (float*)carve((size_t)256 * 513 * 4);
  float* adm   = (float*)carve((size_t)256 * 256 * 4);
  float* q1    = (float*)carve((size_t)256 * 256 * 4);
  float* qA    = (float*)carve((size_t)256 * 256 * 4);
  float* qB    = (float*)carve((size_t)256 * 256 * 4);
  float* kseq  = (float*)carve((size_t)1024 * 256 * 4);
  u16*   gt    = (u16*)carve((size_t)KCOLS * 1024 * 2);   // 33.5 MB (max scale)
  u16*   wt    = (u16*)carve((size_t)256 * KCOLS * 2);    // 8.4 MB
  u16*   ahat  = (u16*)carve((size_t)BE * KCOLS * 2);     // 4.2 MB
  u16*   xdc   = (u16*)carve((size_t)BE * 256 * 2);
  u16*   em    = (u16*)carve((size_t)512 * 256 * 2);
  u16*   f0    = (u16*)carve((size_t)BE * 512 * 2);
  u16*   f1    = (u16*)carve((size_t)BE * 1024 * 2);
  float* dec0  = (float*)carve((size_t)BE * 512 * 4);
  float* dec1  = (float*)carve((size_t)BE * 512 * 4);
  float* meanv = (float*)carve((size_t)BE * 4);
  float* stdv  = (float*)carve((size_t)BE * 4);

  stats_f_kernel<<<BE, 256, 0, stream>>>(x_enc, meanv, stdv, f0, f1);

  for (int s = 0; s < 2; ++s) {
    int L = 512 << s;
    float dt = 1.f / (float)L;

    // --- data-independent constants (rebuilt each call; deterministic) ---
    build_aug_kernel<<<(256 * 513 + 255) / 256, 256, 0, stream>>>(aug, dt);
    gauss_jordan_kernel<<<1, 1024, 0, stream>>>(aug);
    extract_kernel<<<256, 256, 0, stream>>>(aug, adm, q1, kseq);
    seed_kernel<<<1, 256, 0, stream>>>(adm, kseq);        // K[0..32)

    // Q32 = (Ad^T)^32 via 5 squarings on the f32 matrix pipe, ping-pong qA/qB
    gemm_nn_f32_wmma<<<dim3(16, 16), 32, 0, stream>>>(q1, q1, qA, 256, 256, 256);
    float* cur = qA; float* oth = qB;
    for (int t = 1; t < 5; ++t) {
      gemm_nn_f32_wmma<<<dim3(16, 16), 32, 0, stream>>>(cur, cur, oth, 256, 256, 256);
      float* tmp = cur; cur = oth; oth = tmp;
    }
    // doubling: K[d..2d) = K[0..d) * Q_d   (row-major, Q_d = (Ad^d)^T)
    int d = 32;
    while (d < L) {
      gemm_nn_f32_wmma<<<dim3(16, d / 16), 32, 0, stream>>>(
          kseq, cur, kseq + (size_t)d * 256, d, 256, 256);
      d <<= 1;
      if (d < L) {
        gemm_nn_f32_wmma<<<dim3(16, 16), 32, 0, stream>>>(cur, cur, oth, 256, 256, 256);
        float* tmp = cur; cur = oth; oth = tmp;
      }
    }

    em_kernel<<<2, 256, 0, stream>>>(em, L);
    gscan_kernel<<<32, 256, 0, stream>>>(kseq, gt, L);
    wt_kernel<<<256, 256, 0, stream>>>(wre, wim, wt, s, L);

    // --- data GEMMs (bf16 WMMA, f32 accumulate, async-LDS staged B) ---
    const u16* fs = s ? f1 : f0;
    gemm_bf16_nt<<<dim3(KCOLS / 16, BE / (16 * MT)), 32, 0, stream>>>(
        fs, gt, ahat, BE, KCOLS, L, 1);                   // a-hat = f * G
    gemm_bf16_nt<<<dim3(256 / 16, BE / (16 * MT)), 32, 0, stream>>>(
        ahat, wt, xdc, BE, 256, KCOLS, 1);                // xdc = a-hat * W-hat
    gemm_bf16_nt<<<dim3(512 / 16, BE / (16 * MT)), 32, 0, stream>>>(
        xdc, em, s ? dec1 : dec0, BE, 512, 256, 0);       // dec = xdc * Em^T
  }

  final_kernel<<<(4 * 512 * 32) / 256, 256, 0, stream>>>(
      dec0, dec1, mlpw, mlpb, meanv, stdv, out);
}
